// EncoderLSTM_3556232921125
// MI455X (gfx1250) — compile-verified, matrix operand order checked by reading, not run on hardware
//
#include <hip/hip_runtime.h>
#include <hip/hip_bf16.h>

// ---------------------------------------------------------------------------
// LSTM scan for MI455X (gfx1250, wave32, WMMA).
//   SEQ=2048, BATCH=64, D_IN=512, HIDDEN=512, gates i,f,g,o.
// Per-timestep fused kernel: GEMM M=64,N=2048,K=1024 via
// v_wmma_f32_16x16x32_bf16 + gate nonlinearities + c/h update. Weights are
// packed once into WMMA B-operand lane order (bf16) and stay L2-resident
// (4 MB << 192 MB L2). h_prev staging uses the CDNA5 async global->LDS path
// (GLOBAL_LOAD_ASYNC_TO_LDS_B128 + s_wait_asynccnt) to shorten the serial
// critical path between barriers.
// ---------------------------------------------------------------------------

typedef __attribute__((ext_vector_type(16))) __bf16 v16bf;
typedef __attribute__((ext_vector_type(8)))  __bf16 v8bf;
typedef __attribute__((ext_vector_type(8)))  float  v8f;
typedef __attribute__((ext_vector_type(4)))  float  v4f;

#define SEQ_   2048
#define BATCH_ 64
#define DIN_   512
#define HID_   512
#define KTOT_  1024              // DIN + HID (concat [x_t, h])
#define N4H_   2048              // 4*HIDDEN

// ---------------------------------------------------------------------------
// Pack W = [Wi; Wh] (fp32, [1024 x 2048] row-major) into bf16 in the per-lane
// order of the WMMA B operand (32x16 bf16 tile, 16 bf16 contiguous per lane).
// Tile (kt, nt), lane l: column n = nt*16 + (l&15); k offsets within the
// 32-wide K block: base=(l<16?0:8), ks = {base..base+7, base+16..base+23}.
// Storage: vector index v = ((kt*128 + nt)*32 + lane), 16 bf16 each.
// ---------------------------------------------------------------------------
__global__ __launch_bounds__(256) void pack_w_kernel(
    const float* __restrict__ Wi, const float* __restrict__ Wh,
    __bf16* __restrict__ Wp)
{
    int v = blockIdx.x * 256 + threadIdx.x;          // 32*128*32 = 131072
    if (v >= 32 * 128 * 32) return;
    int lane = v & 31;
    int nt   = (v >> 5) & 127;
    int kt   = v >> 12;
    int n    = nt * 16 + (lane & 15);
    int base = (lane < 16) ? 0 : 8;
    __bf16 out[16];
#pragma unroll
    for (int j = 0; j < 16; ++j) {
        int k = kt * 32 + base + j + ((j >= 8) ? 8 : 0);
        float w = (k < DIN_) ? Wi[(size_t)k * N4H_ + n]
                             : Wh[(size_t)(k - DIN_) * N4H_ + n];
        out[j] = (__bf16)w;
    }
    *(v16bf*)(Wp + (size_t)v * 16) = *(const v16bf*)out;
}

__global__ __launch_bounds__(256) void init_h_kernel(
    const float* __restrict__ h0, __bf16* __restrict__ hbuf)
{
    int i = blockIdx.x * 256 + threadIdx.x;
    if (i < BATCH_ * HID_) hbuf[i] = (__bf16)h0[i];
}

// ---------------------------------------------------------------------------
// One LSTM timestep, fully fused.
// Grid: 64 WGs x 256 threads. WG = (wg_m: 32 batch rows) x (wg_n: 16 hidden
// cols, all 4 gates). 8 waves -> 8 output 16x16 tiles (2 M x 4 gates).
// ---------------------------------------------------------------------------
__global__ __launch_bounds__(256) void lstm_step_kernel(
    const float*  __restrict__ x_t,     // [64, 512] fp32 (this timestep)
    const __bf16* __restrict__ h_prev,  // [64, 512] bf16
    const __bf16* __restrict__ Wp,      // packed [1024 x 2048] bf16
    const float*  __restrict__ bias,    // [2048]
    float*        __restrict__ c_state, // [64, 512] fp32 (in/out)
    __bf16*       __restrict__ h_next,  // [64, 512] bf16 (out)
    float*        __restrict__ y_out)   // [64, 512] fp32 slice of ys (out)
{
    extern __shared__ char smem[];
    __bf16* Alds = (__bf16*)smem;                      // [32][512] bf16 (32KB)
    float*  zlds = (float*)(smem + 32 * 512 * 2);      // [32][64]  f32  (8KB)

    const int tid  = threadIdx.x;
    const int wg_n = blockIdx.x & 31;    // hidden slice: cols [wg_n*16, +16)
    const int wg_m = blockIdx.x >> 5;    // batch half:  rows [wg_m*32, +32)
    const int row0 = wg_m * 32;

    const int wave    = tid >> 5;
    const int lane    = tid & 31;
    const int m_local = wave >> 2;            // 0..1
    const int gate    = wave & 3;             // 0..3 (i,f,g,o)
    const int arow    = m_local * 16 + (lane & 15);
    const int kbase   = (lane < 16) ? 0 : 8;
    const int nt      = gate * 32 + wg_n;     // packed N-tile index
    const int jcol    = wg_n * 16 + (lane & 15);

    // Bias folded into the accumulator (constant per output column).
    const float bv = bias[gate * HID_ + jcol];
    v8f acc = {bv, bv, bv, bv, bv, bv, bv, bv};

    // B pointer: advances one full K-block of packed tiles per kt.
    const __bf16* bptr = Wp + ((size_t)nt * 32 + lane) * 16;
    const size_t  bstep = (size_t)128 * 32 * 16;       // one kt of all nt

    // Two K halves: half 0 = x_t (fp32 -> bf16), half 1 = h_prev (bf16).
    for (int half = 0; half < 2; ++half) {
        if (half == 0) {
            // Stage x half: fp32 load -> bf16 convert -> LDS (VGPR path).
            for (int v = tid; v < 32 * 64; v += 256) {
                int r  = v >> 6;
                int k8 = (v & 63) * 8;
                const float* src = x_t + (size_t)(row0 + r) * DIN_ + k8;
                float tmp[8];
                *(v4f*)tmp       = *(const v4f*)src;
                *(v4f*)(tmp + 4) = *(const v4f*)(src + 4);
                v8bf d;
#pragma unroll
                for (int i = 0; i < 8; ++i) d[i] = (__bf16)tmp[i];
                *(v8bf*)(Alds + r * 512 + k8) = d;
            }
        } else {
            // Stage h half: CDNA5 async global->LDS copy, no VGPR bounce.
            // All 8 copies per thread stay in flight (ASYNCcnt tracked).
            for (int v = tid; v < 32 * 64; v += 256) {
                int r  = v >> 6;
                int k8 = (v & 63) * 8;
                unsigned long long ga = (unsigned long long)(const void*)
                    (h_prev + (size_t)(row0 + r) * HID_ + k8);
                unsigned int la = (unsigned int)(unsigned long long)
                    (const void*)(Alds + r * 512 + k8);
                asm volatile("global_load_async_to_lds_b128 %0, %1, off"
                             :: "v"(la), "v"(ga) : "memory");
            }
            asm volatile("s_wait_asynccnt 0x0" ::: "memory");
        }
        __syncthreads();

        const __bf16* aptr = Alds + arow * 512 + kbase;
#pragma unroll 4
        for (int kt = 0; kt < 16; ++kt) {
            v8bf alo = *(const v8bf*)(aptr);
            v8bf ahi = *(const v8bf*)(aptr + 16);
            v16bf a = __builtin_shufflevector(alo, ahi,
                0, 1, 2, 3, 4, 5, 6, 7, 8, 9, 10, 11, 12, 13, 14, 15);
            v16bf b = *(const v16bf*)bptr;
            acc = __builtin_amdgcn_wmma_f32_16x16x32_bf16(
                false, a, false, b, (short)0, acc, false, false);
            aptr += 32;
            bptr += bstep;
        }
        __syncthreads();   // all waves done with this A half before restage
    }

    // Scatter z tile to LDS: C layout -> VGPR r, lane l: row = r + (l>=16?8:0).
    {
        int zr0 = m_local * 16 + ((lane >= 16) ? 8 : 0);
        int zc  = gate * 16 + (lane & 15);
#pragma unroll
        for (int r = 0; r < 8; ++r)
            zlds[(zr0 + r) * 64 + zc] = acc[r];
    }
    __syncthreads();

    // Fused gates + state update: 32 rows x 16 hidden = 512 cells, 2/thread.
    for (int p = tid; p < 512; p += 256) {
        int r = p >> 4;
        int j = p & 15;
        float zi = zlds[r * 64 +  0 + j];
        float zf = zlds[r * 64 + 16 + j];
        float zg = zlds[r * 64 + 32 + j];
        float zo = zlds[r * 64 + 48 + j];
        float ig = 1.f / (1.f + __expf(-zi));
        float fg = 1.f / (1.f + __expf(-zf));
        float e2 = __expf(-2.f * zg);
        float gg = (1.f - e2) / (1.f + e2);
        float og = 1.f / (1.f + __expf(-zo));
        size_t gidx = (size_t)(row0 + r) * HID_ + (wg_n * 16 + j);
        float cn = fg * c_state[gidx] + ig * gg;
        float e2c = __expf(-2.f * cn);
        float hn = og * (1.f - e2c) / (1.f + e2c);
        c_state[gidx] = cn;
        h_next[gidx]  = (__bf16)hn;
        y_out[gidx]   = hn;
    }
}

// ---------------------------------------------------------------------------
extern "C" void kernel_launch(void* const* d_in, const int* in_sizes, int n_in,
                              void* d_out, int out_size, void* d_ws, size_t ws_size,
                              hipStream_t stream)
{
    const float* x  = (const float*)d_in[0];   // [2048, 64, 512]
    const float* c0 = (const float*)d_in[1];   // [64, 512]
    const float* h0 = (const float*)d_in[2];   // [64, 512]
    const float* Wi = (const float*)d_in[3];   // [512, 2048]
    const float* Wh = (const float*)d_in[4];   // [512, 2048]
    const float* b  = (const float*)d_in[5];   // [2048]

    float* out = (float*)d_out;
    float* cT  = out;                           // [64*512]
    float* hT  = out + BATCH_ * HID_;           // [64*512]
    float* ys  = out + 2 * BATCH_ * HID_;       // [2048*64*512]

    // Workspace layout (~4.5 MB):
    char*   ws      = (char*)d_ws;
    __bf16* Wp      = (__bf16*)ws;                                  // 4 MB
    float*  c_state = (float*)(ws + (size_t)4 * 1024 * 1024);       // 128 KB
    __bf16* hbuf0   = (__bf16*)(ws + (size_t)4 * 1024 * 1024 + 128 * 1024);
    __bf16* hbuf1   = hbuf0 + BATCH_ * HID_;                        // 64 KB x2

    // One-time (per launch) prep: pack weights, init states.
    pack_w_kernel<<<(32 * 128 * 32 + 255) / 256, 256, 0, stream>>>(Wi, Wh, Wp);
    init_h_kernel<<<(BATCH_ * HID_ + 255) / 256, 256, 0, stream>>>(h0, hbuf0);
    hipMemcpyAsync(c_state, c0, (size_t)BATCH_ * HID_ * sizeof(float),
                   hipMemcpyDeviceToDevice, stream);

    const size_t lds_bytes = 32 * 512 * 2 + 32 * 64 * 4;   // 40960 B

    for (int t = 0; t < SEQ_; ++t) {
        const __bf16* hp = (t & 1) ? hbuf1 : hbuf0;
        __bf16*       hn = (t & 1) ? hbuf0 : hbuf1;
        lstm_step_kernel<<<64, 256, lds_bytes, stream>>>(
            x + (size_t)t * BATCH_ * DIN_, hp, Wp, b, c_state, hn,
            ys + (size_t)t * BATCH_ * HID_);
    }

    hipMemcpyAsync(cT, c_state, (size_t)BATCH_ * HID_ * sizeof(float),
                   hipMemcpyDeviceToDevice, stream);
    hipMemcpyAsync(hT, ys + (size_t)(SEQ_ - 1) * BATCH_ * HID_,
                   (size_t)BATCH_ * HID_ * sizeof(float),
                   hipMemcpyDeviceToDevice, stream);
}
